// LatentSDE_30434138260127
// MI455X (gfx1250) — compile-verified, matrix operand order checked by reading, not run on hardware
//
#include <hip/hip_runtime.h>
#include <hip/hip_bf16.h>

#define T_STEPS 100
#define BATCH   4096
#define DDIM    64
#define CDIM    64
#define HDIM    128
#define BM      64          // batch rows per workgroup
#define NTHREADS 256        // 8 wave32s

typedef __attribute__((ext_vector_type(16))) _Float16 v16h;
typedef __attribute__((ext_vector_type(8)))  _Float16 v8h;
typedef __attribute__((ext_vector_type(8)))  float    v8f;
typedef __attribute__((ext_vector_type(4)))  float    v4f;

// ---- WMMA helpers -----------------------------------------------------------
// A-fragment (16x32 f16): lane L holds row M=L&15; halves 0..7 = K kb..kb+7,
// halves 8..15 = K kb+16..kb+23 with kb = (L>=16)?8:0.
// B-fragment (32x16 f16) from row-major W(out,in): lane L holds column N=L&15
// (= W row), same K striping along W's input dim.  Both reduce to two 16-byte
// loads per lane from a row-major tile with leading dimension `ld`.
__device__ __forceinline__ v16h ldfrag(const _Float16* __restrict__ base,
                                       int ld, int rbase, int kofs) {
  const int lane = threadIdx.x & 31;
  const _Float16* p = base + (rbase + (lane & 15)) * ld + kofs + ((lane >> 4) << 3);
  v8h lo = *(const v8h*)(p);
  v8h hi = *(const v8h*)(p + 16);
  v16h r;
#pragma unroll
  for (int i = 0; i < 8; ++i) { r[i] = lo[i]; r[8 + i] = hi[i]; }
  return r;
}

__device__ __forceinline__ v8f wmma_f16(v16h a, v16h b, v8f c) {
  return __builtin_amdgcn_wmma_f32_16x16x32_f16(false, a, false, b, (short)0, c,
                                                false, false);
}

__device__ __forceinline__ float sp_f(float x) {       // softplus
  return (x > 20.f) ? x : __logf(1.f + __expf(x));
}
__device__ __forceinline__ float sig_f(float x) {      // sigmoid
  return 1.f / (1.f + __expf(-x));
}

// ---- f32 -> f16 weight pack -------------------------------------------------
__global__ void cvt_f16_kernel(const float* __restrict__ src,
                               _Float16* __restrict__ dst, int n) {
  int i = blockIdx.x * NTHREADS + threadIdx.x;
  if (i < n) dst[i] = (_Float16)src[i];
}

// ---- Phase 1: backward GRU + encoder -> ctx(T,B,C) --------------------------
__global__ __launch_bounds__(NTHREADS)
void gru_enc_kernel(const float* __restrict__ xs,
                    const float* __restrict__ bih, const float* __restrict__ bhh,
                    const float* __restrict__ enc_b,
                    const _Float16* __restrict__ wih16,   // (384,64)
                    const _Float16* __restrict__ whh16,   // (384,128)
                    const _Float16* __restrict__ enc16,   // (64,128)
                    float* __restrict__ ctx) {
  __shared__ _Float16 sx[BM * DDIM];          // x_t tile, f16
  __shared__ _Float16 sh[2][BM * HDIM];       // double-buffered hidden state, f16

  const int tid  = threadIdx.x;
  const int lane = tid & 31;
  const int wave = tid >> 5;
  const int rowbase = blockIdx.x * BM;

  for (int i = tid; i < BM * HDIM; i += NTHREADS) sh[0][i] = (_Float16)0.f;

  const int j = wave * 16 + (lane & 15);      // owned column within H
  const float br  = bih[j]            + bhh[j];
  const float bz  = bih[HDIM + j]     + bhh[HDIM + j];
  const float bin = bih[2 * HDIM + j];
  const float bhn = bhh[2 * HDIM + j];

  int cur = 0;
  for (int t = 0; t < T_STEPS; ++t) {
    const int tr = T_STEPS - 1 - t;           // reversed scan
    __syncthreads();
    const float* xsrc = xs + ((size_t)tr * BATCH + rowbase) * DDIM;
    for (int i = tid * 4; i < BM * DDIM; i += NTHREADS * 4) {
      v4f v = *(const v4f*)(xsrc + i);
      sx[i + 0] = (_Float16)v.x; sx[i + 1] = (_Float16)v.y;
      sx[i + 2] = (_Float16)v.z; sx[i + 3] = (_Float16)v.w;
    }
    if (tr > 0)
      __builtin_prefetch(xs + ((size_t)(tr - 1) * BATCH + rowbase) * DDIM + tid * 16, 0, 3);
    __syncthreads();

    const int nxt = cur ^ 1;
    const int jb  = wave * 16;
#pragma unroll 1
    for (int mt = 0; mt < BM / 16; ++mt) {
      v16h ax0 = ldfrag(sx, DDIM, mt * 16, 0);
      v16h ax1 = ldfrag(sx, DDIM, mt * 16, 32);
      v16h ah0 = ldfrag(sh[cur], HDIM, mt * 16, 0);
      v16h ah1 = ldfrag(sh[cur], HDIM, mt * 16, 32);
      v16h ah2 = ldfrag(sh[cur], HDIM, mt * 16, 64);
      v16h ah3 = ldfrag(sh[cur], HDIM, mt * 16, 96);

      v8f ar = {}; v8f az = {}; v8f an = {}; v8f ah = {};
      // r gate: x@Wih_r + h@Whh_r
      ar = wmma_f16(ax0, ldfrag(wih16, DDIM, jb, 0),  ar);
      ar = wmma_f16(ax1, ldfrag(wih16, DDIM, jb, 32), ar);
      ar = wmma_f16(ah0, ldfrag(whh16, HDIM, jb, 0),  ar);
      ar = wmma_f16(ah1, ldfrag(whh16, HDIM, jb, 32), ar);
      ar = wmma_f16(ah2, ldfrag(whh16, HDIM, jb, 64), ar);
      ar = wmma_f16(ah3, ldfrag(whh16, HDIM, jb, 96), ar);
      // z gate
      az = wmma_f16(ax0, ldfrag(wih16, DDIM, HDIM + jb, 0),  az);
      az = wmma_f16(ax1, ldfrag(wih16, DDIM, HDIM + jb, 32), az);
      az = wmma_f16(ah0, ldfrag(whh16, HDIM, HDIM + jb, 0),  az);
      az = wmma_f16(ah1, ldfrag(whh16, HDIM, HDIM + jb, 32), az);
      az = wmma_f16(ah2, ldfrag(whh16, HDIM, HDIM + jb, 64), az);
      az = wmma_f16(ah3, ldfrag(whh16, HDIM, HDIM + jb, 96), az);
      // inn (x part of n gate)
      an = wmma_f16(ax0, ldfrag(wih16, DDIM, 2 * HDIM + jb, 0),  an);
      an = wmma_f16(ax1, ldfrag(wih16, DDIM, 2 * HDIM + jb, 32), an);
      // hn (h part of n gate, kept separate for r*hn)
      ah = wmma_f16(ah0, ldfrag(whh16, HDIM, 2 * HDIM + jb, 0),  ah);
      ah = wmma_f16(ah1, ldfrag(whh16, HDIM, 2 * HDIM + jb, 32), ah);
      ah = wmma_f16(ah2, ldfrag(whh16, HDIM, 2 * HDIM + jb, 64), ah);
      ah = wmma_f16(ah3, ldfrag(whh16, HDIM, 2 * HDIM + jb, 96), ah);

#pragma unroll
      for (int i = 0; i < 8; ++i) {
        const int row = mt * 16 + ((lane >> 4) << 3) + i;
        float r = sig_f(ar[i] + br);
        float z = sig_f(az[i] + bz);
        float n = tanhf(an[i] + bin + r * (ah[i] + bhn));
        float hold = (float)sh[cur][row * HDIM + j];
        sh[nxt][row * HDIM + j] = (_Float16)((1.f - z) * n + z * hold);
      }
    }
    __syncthreads();
    // encoder: ctx[tr] = h_new @ encW^T + enc_b
    float* cdst = ctx + ((size_t)tr * BATCH + rowbase) * CDIM;
#pragma unroll
    for (int e = 0; e < 2; ++e) {
      const int tt = wave * 2 + e;
      const int mt = tt >> 2, nt = tt & 3;
      v8f acc = {};
      acc = wmma_f16(ldfrag(sh[nxt], HDIM, mt * 16, 0),  ldfrag(enc16, HDIM, nt * 16, 0),  acc);
      acc = wmma_f16(ldfrag(sh[nxt], HDIM, mt * 16, 32), ldfrag(enc16, HDIM, nt * 16, 32), acc);
      acc = wmma_f16(ldfrag(sh[nxt], HDIM, mt * 16, 64), ldfrag(enc16, HDIM, nt * 16, 64), acc);
      acc = wmma_f16(ldfrag(sh[nxt], HDIM, mt * 16, 96), ldfrag(enc16, HDIM, nt * 16, 96), acc);
      const int col = nt * 16 + (lane & 15);
      const float eb = enc_b[col];
#pragma unroll
      for (int i = 0; i < 8; ++i) {
        const int row = mt * 16 + ((lane >> 4) << 3) + i;
        cdst[row * CDIM + col] = acc[i] + eb;
      }
    }
    cur = nxt;
  }
}

// ---- Phase 2: Euler–Maruyama SDE loop with logqp ----------------------------
__global__ __launch_bounds__(NTHREADS)
void sde_kernel(const float* __restrict__ xs, const float* __restrict__ ts,
                const float* __restrict__ dw, const float* __restrict__ noise_std,
                const float* __restrict__ ctx,
                const _Float16* __restrict__ fw1, const float* __restrict__ fb1,
                const _Float16* __restrict__ fw2, const float* __restrict__ fb2,
                const _Float16* __restrict__ fw3, const float* __restrict__ fb3,
                const _Float16* __restrict__ hw1, const float* __restrict__ hb1,
                const _Float16* __restrict__ hw2, const float* __restrict__ hb2,
                const _Float16* __restrict__ hw3, const float* __restrict__ hb3,
                const float* __restrict__ gW1, const float* __restrict__ gb1,
                const float* __restrict__ gW2, const float* __restrict__ gb2,
                float* __restrict__ lp_row, float* __restrict__ lr_row) {
  __shared__ float    sy[BM * DDIM];              // y state, f32
  __shared__ _Float16 A0[BM * HDIM];              // [y|c], later l2
  __shared__ _Float16 A1[BM * HDIM];              // f-net l1
  __shared__ _Float16 A2[BM * HDIM];              // h-net a1
  __shared__ _Float16 A3[BM * HDIM];              // h-net a2
  __shared__ float    slp[BM], slr[BM];

  const int tid  = threadIdx.x;
  const int lane = tid & 31;
  const int wave = tid >> 5;
  const int rowbase = blockIdx.x * BM;

  const float sigma   = noise_std[0];
  const float inv_std = 1.f / sigma;
  const float lpc     = -__logf(sigma) - 0.9189385332f;   // -log s - 0.5*log(2*pi)

  // y0 = xs[0]
  for (int i = tid * 4; i < BM * DDIM; i += NTHREADS * 4) {
    v4f v = *(const v4f*)(xs + (size_t)rowbase * DDIM + i);
    sy[i] = v.x; sy[i + 1] = v.y; sy[i + 2] = v.z; sy[i + 3] = v.w;
  }
  for (int i = tid; i < BM; i += NTHREADS) { slp[i] = 0.f; slr[i] = 0.f; }

  const int jb = wave * 16;
  const int j  = jb + (lane & 15);
  const float b1f = fb1[j], b1h = hb1[j], b2f = fb2[j], b2h = hb2[j];

#pragma unroll 1
  for (int t = 0; t < T_STEPS - 1; ++t) {
    const float dtk  = ts[t + 1] - ts[t];
    const float sqdt = sqrtf(dtk);
    __syncthreads();
    // A0 = [y | ctx[t+1]] as f16
    const float* csrc = ctx + ((size_t)(t + 1) * BATCH + rowbase) * CDIM;
    for (int idx = tid; idx < BM * DDIM; idx += NTHREADS) {
      const int row = idx >> 6, d = idx & 63;
      A0[row * HDIM + d]        = (_Float16)sy[idx];
      A0[row * HDIM + CDIM + d] = (_Float16)csrc[idx];
    }
    __syncthreads();

    // layer 1: l1 = sp([y|c]@fW1^T+fb1) -> A1 ; a1 = sp(y@hW1^T+hb1) -> A2
#pragma unroll 1
    for (int mt = 0; mt < BM / 16; ++mt) {
      v16h a0 = ldfrag(A0, HDIM, mt * 16, 0);
      v16h a1 = ldfrag(A0, HDIM, mt * 16, 32);
      v16h a2 = ldfrag(A0, HDIM, mt * 16, 64);
      v16h a3 = ldfrag(A0, HDIM, mt * 16, 96);
      v8f af = {}; v8f ah = {};
      af = wmma_f16(a0, ldfrag(fw1, HDIM, jb, 0),  af);
      af = wmma_f16(a1, ldfrag(fw1, HDIM, jb, 32), af);
      af = wmma_f16(a2, ldfrag(fw1, HDIM, jb, 64), af);
      af = wmma_f16(a3, ldfrag(fw1, HDIM, jb, 96), af);
      ah = wmma_f16(a0, ldfrag(hw1, DDIM, jb, 0),  ah);   // y-only, K=64
      ah = wmma_f16(a1, ldfrag(hw1, DDIM, jb, 32), ah);
#pragma unroll
      for (int i = 0; i < 8; ++i) {
        const int row = mt * 16 + ((lane >> 4) << 3) + i;
        A1[row * HDIM + j] = (_Float16)sp_f(af[i] + b1f);
        A2[row * HDIM + j] = (_Float16)sp_f(ah[i] + b1h);
      }
    }
    __syncthreads();

    // layer 2: l2 = sp(l1@fW2^T+fb2) -> A0 ; a2 = sp(a1@hW2^T+hb2) -> A3
#pragma unroll 1
    for (int mt = 0; mt < BM / 16; ++mt) {
      v8f af = {}; v8f ah = {};
#pragma unroll
      for (int kk = 0; kk < 4; ++kk) {
        v16h fa = ldfrag(A1, HDIM, mt * 16, kk * 32);
        v16h ha = ldfrag(A2, HDIM, mt * 16, kk * 32);
        af = wmma_f16(fa, ldfrag(fw2, HDIM, jb, kk * 32), af);
        ah = wmma_f16(ha, ldfrag(hw2, HDIM, jb, kk * 32), ah);
      }
#pragma unroll
      for (int i = 0; i < 8; ++i) {
        const int row = mt * 16 + ((lane >> 4) << 3) + i;
        A0[row * HDIM + j] = (_Float16)sp_f(af[i] + b2f);
        A3[row * HDIM + j] = (_Float16)sp_f(ah[i] + b2h);
      }
    }
    __syncthreads();

    // layer 3 + g-net + SDE update + log-prob; each wave owns two matched
    // 16x16 tiles of fy and hy so u=(f-h)/g is wave-local.
    const float* dwsrc = dw + ((size_t)t * BATCH + rowbase) * DDIM;
    const float* xnext = xs + ((size_t)(t + 1) * BATCH + rowbase) * DDIM;
#pragma unroll 1
    for (int e = 0; e < 2; ++e) {
      const int tt = wave * 2 + e;
      const int mt = tt >> 2, nt = tt & 3;
      v8f fy = {}; v8f hy = {};
#pragma unroll
      for (int kk = 0; kk < 4; ++kk) {
        fy = wmma_f16(ldfrag(A0, HDIM, mt * 16, kk * 32),
                      ldfrag(fw3, HDIM, nt * 16, kk * 32), fy);
        hy = wmma_f16(ldfrag(A3, HDIM, mt * 16, kk * 32),
                      ldfrag(hw3, HDIM, nt * 16, kk * 32), hy);
      }
      const int d = nt * 16 + (lane & 15);
      const float fb = fb3[d], hb = hb3[d], gb = gb2[d];
      const float* g1p = gW1 + d * HDIM;
      const float* gbp = gb1 + d * HDIM;
      const float* g2p = gW2 + d * HDIM;

      float yv[8], gacc[8];
#pragma unroll
      for (int i = 0; i < 8; ++i) {
        const int row = mt * 16 + ((lane >> 4) << 3) + i;
        yv[i] = sy[row * DDIM + d];
        gacc[i] = 0.f;
      }
#pragma unroll 4
      for (int hh = 0; hh < HDIM; ++hh) {          // tiny per-dim g-net
        const float w1 = g1p[hh], bb = gbp[hh], w2 = g2p[hh];
#pragma unroll
        for (int i = 0; i < 8; ++i) gacc[i] += sp_f(fmaf(yv[i], w1, bb)) * w2;
      }
#pragma unroll
      for (int i = 0; i < 8; ++i) {
        const int row = mt * 16 + ((lane >> 4) << 3) + i;
        const float f = fy[i] + fb;
        const float h = hy[i] + hb;
        const float g = sig_f(gacc[i] + gb);
        const float u = (f - h) / g;
        atomicAdd(&slr[row], 0.5f * u * u * dtk);
        const float y2 = yv[i] + f * dtk + g * sqdt * dwsrc[row * DDIM + d];
        const float df = (xnext[row * DDIM + d] - y2) * inv_std;
        atomicAdd(&slp[row], -0.5f * df * df + lpc);
        sy[row * DDIM + d] = y2;                   // (row,d) owned by this lane
      }
    }
  }
  __syncthreads();
  for (int i = tid; i < BM; i += NTHREADS) {
    lp_row[rowbase + i] = slp[i];
    lr_row[rowbase + i] = slr[i];
  }
}

// ---- Phase 3: reduce to the two scalars -------------------------------------
__global__ __launch_bounds__(NTHREADS)
void reduce_kernel(const float* __restrict__ lp_row, const float* __restrict__ lr_row,
                   const float* __restrict__ noise_std, float* __restrict__ out) {
  __shared__ float s0[NTHREADS], s1[NTHREADS];
  float a = 0.f, b = 0.f;
  for (int i = threadIdx.x; i < BATCH; i += NTHREADS) { a += lp_row[i]; b += lr_row[i]; }
  s0[threadIdx.x] = a; s1[threadIdx.x] = b;
  __syncthreads();
  for (int k = NTHREADS / 2; k > 0; k >>= 1) {
    if (threadIdx.x < k) { s0[threadIdx.x] += s0[threadIdx.x + k];
                           s1[threadIdx.x] += s1[threadIdx.x + k]; }
    __syncthreads();
  }
  if (threadIdx.x == 0) {
    const float sigma = noise_std[0];
    const float lpc   = -__logf(sigma) - 0.9189385332f;
    out[0] = s0[0] / (float)BATCH + (float)DDIM * lpc;   // t=0 term: diff==0
    out[1] = s1[0] / (float)BATCH;
  }
}

// ---- host launch ------------------------------------------------------------
extern "C" void kernel_launch(void* const* d_in, const int* in_sizes, int n_in,
                              void* d_out, int out_size, void* d_ws, size_t ws_size,
                              hipStream_t stream) {
  (void)in_sizes; (void)n_in; (void)out_size; (void)ws_size;
  const float* xs        = (const float*)d_in[0];
  const float* ts        = (const float*)d_in[1];
  const float* dw        = (const float*)d_in[2];
  const float* noise_std = (const float*)d_in[3];
  const float* gru_bih   = (const float*)d_in[5];
  const float* gru_bhh   = (const float*)d_in[7];
  const float* enc_b     = (const float*)d_in[9];
  const float* fb1 = (const float*)d_in[11];
  const float* fb2 = (const float*)d_in[13];
  const float* fb3 = (const float*)d_in[15];
  const float* hb1 = (const float*)d_in[17];
  const float* hb2 = (const float*)d_in[19];
  const float* hb3 = (const float*)d_in[21];
  const float* gW1 = (const float*)d_in[22];
  const float* gb1 = (const float*)d_in[23];
  const float* gW2 = (const float*)d_in[24];
  const float* gb2 = (const float*)d_in[25];

  char* ws = (char*)d_ws;
  const size_t CTX_BYTES = (size_t)T_STEPS * BATCH * CDIM * sizeof(float);
  float*    ctx    = (float*)ws;
  float*    lp_row = (float*)(ws + CTX_BYTES);
  float*    lr_row = lp_row + BATCH;
  _Float16* w16    = (_Float16*)(ws + CTX_BYTES + 2 * (size_t)BATCH * sizeof(float));

  // f16 weight pack offsets (in halves)
  const size_t oWih = 0,      oWhh = 24576, oEnc = 73728;
  const size_t oF1  = 81920,  oF2  = 98304, oF3  = 114688;
  const size_t oH1  = 122880, oH2  = 131072, oH3 = 147456;
  const void*  wsrc[9] = { d_in[4], d_in[6], d_in[8], d_in[10], d_in[12],
                           d_in[14], d_in[16], d_in[18], d_in[20] };
  const int    wn[9]   = { 24576, 49152, 8192, 16384, 16384, 8192, 8192, 16384, 8192 };
  const size_t wo[9]   = { oWih, oWhh, oEnc, oF1, oF2, oF3, oH1, oH2, oH3 };
  for (int i = 0; i < 9; ++i)
    cvt_f16_kernel<<<(wn[i] + NTHREADS - 1) / NTHREADS, NTHREADS, 0, stream>>>(
        (const float*)wsrc[i], w16 + wo[i], wn[i]);

  gru_enc_kernel<<<BATCH / BM, NTHREADS, 0, stream>>>(
      xs, gru_bih, gru_bhh, enc_b, w16 + oWih, w16 + oWhh, w16 + oEnc, ctx);

  sde_kernel<<<BATCH / BM, NTHREADS, 0, stream>>>(
      xs, ts, dw, noise_std, ctx,
      w16 + oF1, fb1, w16 + oF2, fb2, w16 + oF3, fb3,
      w16 + oH1, hb1, w16 + oH2, hb2, w16 + oH3, hb3,
      gW1, gb1, gW2, gb2, lp_row, lr_row);

  reduce_kernel<<<1, NTHREADS, 0, stream>>>(lp_row, lr_row, noise_std, (float*)d_out);
}
